// Decoder_65644280152151
// MI455X (gfx1250) — matrix operand order repaired and checked by reading, not compiled, not run on hardware
//
#include <hip/hip_runtime.h>
#include <math.h>

// ---------------- WMMA / TDM types ----------------
typedef __attribute__((ext_vector_type(16))) __bf16        v16bf;
typedef __attribute__((ext_vector_type(8)))  float         v8f;
typedef __attribute__((ext_vector_type(4)))  unsigned int  u32x4;
typedef __attribute__((ext_vector_type(8)))  unsigned int  u32x8;

#define B_  64
#define S_  400
#define H_  512
#define V_  32000

// TDM-padded LDS tile: 16 rows x 1024 f32, +2 DWORDs pad per 256 DWORDs
// => effective row stride 1032 DWORDs, total 16*1024 + 128 = 16512 DWORDs (64.5 KB)
#define TILE_LDS_DW 16512

// ----- fragment loaders -----
// A fragment (16x32 MxK) from fp32 global: lane row = lane&15,
// per-lane K = k0 + 16*hk + e
__device__ __forceinline__ v16bf load_a_frag(const float* __restrict__ base /* A + row*K + k0 */, int hk) {
    v16bf a;
    const float* p = base + 16 * hk;
#pragma unroll
    for (int i = 0; i < 8; ++i) {
        a[i]     = (__bf16)p[i];
        a[8 + i] = (__bf16)p[8 + i];
    }
    return a;
}
// A fragment from the TDM-padded LDS tile (pad: +2 DW per 256 DW stored)
__device__ __forceinline__ v16bf load_a_frag_lds(const float* s_a, int lr, int k0, int hk) {
    const int s1 = k0 + 16 * hk;       // 16-aligned, never crosses a 256-DW pad boundary
    const int s2 = s1 + 8;
    const float* p1 = s_a + lr * 1032 + s1 + 2 * (s1 >> 8);
    const float* p2 = s_a + lr * 1032 + s2 + 2 * (s2 >> 8);
    v16bf a;
#pragma unroll
    for (int i = 0; i < 8; ++i) {
        a[i]     = (__bf16)p1[i];
        a[8 + i] = (__bf16)p2[i];
    }
    return a;
}
// B fragment (32x16 KxN) from W[N,K] row-major fp32: lane col = lane&15, K = k0 + 16*hk + e
__device__ __forceinline__ v16bf load_b_frag(const float* __restrict__ base /* W + n*K + k0 */, int hk) {
    v16bf b;
    const float* p = base + 16 * hk;
#pragma unroll
    for (int e = 0; e < 16; ++e) b[e] = (__bf16)p[e];
    return b;
}
// B fragment straight from pre-converted bf16 weights (32B-aligned vector load, no cvt)
__device__ __forceinline__ v16bf load_b_frag_bf16(const __bf16* __restrict__ base /* W + n*K + k0 */, int hk) {
    return *(const v16bf*)(base + 16 * hk);
}

// ---------------- fp32 -> bf16 conversion (weights prepass) ----------------
__global__ void f32_to_bf16_kernel(const float* __restrict__ src, __bf16* __restrict__ dst, int n) {
    int i = blockIdx.x * blockDim.x + threadIdx.x;
    if (i < n) dst[i] = (__bf16)src[i];
}

// ---------------- Generic NT GEMM: C[M,N] = A[M,K] @ W[N,K]^T + bias ----------------
// 256 threads = 8 waves; each wave owns one 16x16 tile. grid = (ceil(N/128), M/16).
__global__ void gemm_nt_bf16_wmma(const float* __restrict__ A, const float* __restrict__ W,
                                  const float* __restrict__ bias, float* __restrict__ C,
                                  int M, int N, int K) {
    const int lane = threadIdx.x & 31;
    const int wv   = threadIdx.x >> 5;
    const int hk   = lane >> 4;
    const int lr   = lane & 15;
    const int n0   = (blockIdx.x * 8 + wv) * 16;
    if (n0 >= N) return;                      // wave-uniform: EXEC stays all-ones for WMMA
    const int m0   = blockIdx.y * 16;

    const float* arow = A + (long)(m0 + lr) * K;
    const float* wrow = W + (long)(n0 + lr) * K;

    v8f acc = {};
    for (int k0 = 0; k0 < K; k0 += 32) {
        if (k0 + 128 < K) __builtin_prefetch(wrow + k0 + 128, 0, 0);  // global_prefetch_b8 on W stream
        v16bf a = load_a_frag(arow + k0, hk);
        v16bf b = load_b_frag(wrow + k0, hk);
        acc = __builtin_amdgcn_wmma_f32_16x16x32_bf16(false, a, false, b, (short)0, acc, false, false);
    }
    const int col = n0 + lr;
    const float bv = bias ? bias[col] : 0.0f;
#pragma unroll
    for (int j = 0; j < 8; ++j) {
        const int row = m0 + j + 8 * hk;
        C[(long)row * N + col] = acc[j] + bv;
    }
}

// ---------------- GRU cell elementwise (gates already have biases) ----------------
__global__ void gru_cell_kernel(const float* __restrict__ gi, const float* __restrict__ gh,
                                const float* __restrict__ hprev,
                                float* __restrict__ hidden_out, float* __restrict__ concat) {
    int idx = blockIdx.x * blockDim.x + threadIdx.x;
    if (idx >= B_ * H_) return;
    int b = idx >> 9, h = idx & (H_ - 1);
    const float* gib = gi + (long)b * 3 * H_;
    const float* ghb = gh + (long)b * 3 * H_;
    float r = 1.0f / (1.0f + expf(-(gib[h] + ghb[h])));
    float z = 1.0f / (1.0f + expf(-(gib[H_ + h] + ghb[H_ + h])));
    float n = tanhf(gib[2 * H_ + h] + r * ghb[2 * H_ + h]);
    float hv = (1.0f - z) * n + z * hprev[idx];
    hidden_out[idx] = hv;
    concat[(long)b * (3 * H_) + h] = hv;   // concat row layout: [hidden(512) | context(1024)]
}

// ---------------- Fused attention scores (TDM-staged A tile) ----------------
// scores[m] = sum_n vatt[n]*tanh( (enc@Wh^T)[m,n] + Wh_b[n] + sproj[b,n] ),  m = b*S + s
// One workgroup per 16-row tile (S=400 = 25*16 -> tile stays inside one batch).
// Wave 0 issues one tensor_load_to_lds for the 16x1024 fp32 A tile (TDM pad rotates
// LDS banks by 8 DW/row); 8 waves x 4 col-tiles cover all 512 N columns with WMMA,
// B fragments come from pre-converted bf16 weights.
__global__ void attn_scores_wmma(const float* __restrict__ enc,    /* [B*S, 1024] f32 */
                                 const __bf16* __restrict__ Whbf,  /* [512, 1024] bf16 */
                                 const float* __restrict__ Wh_b,   /* [512] */
                                 const float* __restrict__ sproj,  /* [B, 512] */
                                 const float* __restrict__ vatt,   /* [512] */
                                 float* __restrict__ scores /* [B*S] */) {
    const int lane = threadIdx.x & 31;
    const int wv   = threadIdx.x >> 5;
    const int hk   = lane >> 4;
    const int lr   = lane & 15;
    const int m0   = blockIdx.x * 16;
    const int b    = m0 / S_;

    __shared__ float s_a[TILE_LDS_DW];   // TDM destination (padded 16x1024 f32)
    __shared__ float s_acc[16];

    // ---- wave 0: build D# and launch the Tensor Data Mover ----
    if (wv == 0) {
        unsigned long long ga = (unsigned long long)(uintptr_t)(enc + (size_t)m0 * 1024);
        unsigned lds_off = (unsigned)(uintptr_t)(&s_a[0]);   // generic LDS addr: low 32 bits = LDS offset

        u32x4 g0;
        g0[0] = 1u;                                              // count=1 valid descriptor
        g0[1] = lds_off;                                         // lds_addr
        g0[2] = (unsigned)ga;                                    // global_addr[31:0]
        g0[3] = (unsigned)((ga >> 32) & 0x1FFFFFFull)            // global_addr[56:32]
              | (2u << 30);                                      // type=2 ("image")

        u32x8 g1;
        g1[0] = (2u << 16)                                       // data_size=4B
              | (1u << 20)                                       // pad_enable
              | (7u << 22)                                       // pad_interval: 256 DWORDs
              | (1u << 25);                                      // pad_amount: 2 DWORDs
        g1[1] = (1024u & 0xFFFFu) << 16;                         // tensor_dim0[15:0]=1024
        g1[2] = (1024u >> 16) | (16u << 16);                     // tensor_dim0[31:16] | tensor_dim1[15:0]=16
        g1[3] = (1024u << 16);                                   // tensor_dim1[31:16]=0 | tile_dim0=1024
        g1[4] = 16u;                                             // tile_dim1=16, tile_dim2=0
        g1[5] = 1024u;                                           // tensor_dim0_stride[31:0]
        g1[6] = 0u;                                              // stride hi | tensor_dim1_stride lo
        g1[7] = 0u;

        asm volatile("tensor_load_to_lds %0, %1" :: "s"(g0), "s"(g1) : "memory");
    }
    if (threadIdx.x < 16) s_acc[threadIdx.x] = 0.0f;

    __builtin_amdgcn_s_wait_tensorcnt(0);   // no-op for non-issuing waves (TENSORcnt==0)
    __syncthreads();                        // LDS tile + s_acc visible to all waves

    const int nbase = wv * 64;  // this wave's 4 tiles: cols [nbase, nbase+64)

    v8f acc[4];
#pragma unroll
    for (int t = 0; t < 4; ++t) acc[t] = (v8f){};

    for (int k0 = 0; k0 < 2 * H_; k0 += 32) {
        v16bf a = load_a_frag_lds(s_a, lr, k0, hk);
#pragma unroll
        for (int t = 0; t < 4; ++t) {
            const __bf16* wrow = Whbf + (long)(nbase + t * 16 + lr) * (2 * H_);
            v16bf bf = load_b_frag_bf16(wrow + k0, hk);
            acc[t] = __builtin_amdgcn_wmma_f32_16x16x32_bf16(false, a, false, bf, (short)0, acc[t], false, false);
        }
    }

#pragma unroll
    for (int t = 0; t < 4; ++t) {
        const int n = nbase + t * 16 + lr;
        const float addv = Wh_b[n] + sproj[(long)b * H_ + n];
        const float vn = vatt[n];
#pragma unroll
        for (int j = 0; j < 8; ++j) {
            float val = tanhf(acc[t][j] + addv) * vn;
            // sum over the 16 columns of this tile (lanes sharing hk)
            val += __shfl_xor(val, 1, 32);
            val += __shfl_xor(val, 2, 32);
            val += __shfl_xor(val, 4, 32);
            val += __shfl_xor(val, 8, 32);
            if (lr == 0) atomicAdd(&s_acc[j + 8 * hk], val);
        }
    }
    __syncthreads();
    if (threadIdx.x < 16) scores[m0 + threadIdx.x] = s_acc[threadIdx.x];
}

// ---------------- Masked softmax over S (one block per batch row) ----------------
__global__ void softmax_attn_kernel(const float* __restrict__ scores,
                                    const unsigned char* __restrict__ mask, /* jax bool = 1 byte */
                                    float* __restrict__ attn) {
    const int b = blockIdx.x, tid = threadIdx.x;
    __shared__ float red[256];
    float mx = -INFINITY;
    for (int i = tid; i < S_; i += 256) {
        float v = mask[b * S_ + i] ? scores[b * S_ + i] : -INFINITY;
        mx = fmaxf(mx, v);
    }
    red[tid] = mx; __syncthreads();
    for (int s = 128; s > 0; s >>= 1) { if (tid < s) red[tid] = fmaxf(red[tid], red[tid + s]); __syncthreads(); }
    mx = red[0]; __syncthreads();

    float sm = 0.0f;
    for (int i = tid; i < S_; i += 256)
        sm += mask[b * S_ + i] ? expf(scores[b * S_ + i] - mx) : 0.0f;
    red[tid] = sm; __syncthreads();
    for (int s = 128; s > 0; s >>= 1) { if (tid < s) red[tid] += red[tid + s]; __syncthreads(); }
    const float inv = 1.0f / red[0];

    for (int i = tid; i < S_; i += 256) {
        float v = mask[b * S_ + i] ? expf(scores[b * S_ + i] - mx) : 0.0f;
        attn[b * S_ + i] = v * inv;
    }
}

// ---------------- Context: ctx[b,d] = sum_s attn[b,s] * enc[b,s,d] ----------------
__global__ void context_kernel(const float* __restrict__ attn, const float* __restrict__ enc,
                               float* __restrict__ ctx_out, float* __restrict__ concat) {
    const int b = blockIdx.x, tid = threadIdx.x;
    __shared__ float s_attn[S_];
    for (int i = tid; i < S_; i += 256) s_attn[i] = attn[b * S_ + i];
    __syncthreads();
    for (int d = tid; d < 2 * H_; d += 256) {
        const float* e = enc + (long)b * S_ * (2 * H_) + d;
        float sum = 0.0f;
        for (int s = 0; s < S_; ++s) sum += s_attn[s] * e[(long)s * (2 * H_)];
        ctx_out[(long)b * (2 * H_) + d] = sum;
        concat[(long)b * (3 * H_) + H_ + d] = sum;
    }
}

// ---------------- Vocab softmax (one block per batch row, V=32000) ----------------
__global__ void softmax_vocab_kernel(const float* __restrict__ logits, float* __restrict__ vocab) {
    const int b = blockIdx.x, tid = threadIdx.x;
    __shared__ float red[256];
    const float* row = logits + (long)b * V_;
    float mx = -INFINITY;
    for (int i = tid; i < V_; i += 256) mx = fmaxf(mx, row[i]);
    red[tid] = mx; __syncthreads();
    for (int s = 128; s > 0; s >>= 1) { if (tid < s) red[tid] = fmaxf(red[tid], red[tid + s]); __syncthreads(); }
    mx = red[0]; __syncthreads();

    float sm = 0.0f;
    for (int i = tid; i < V_; i += 256) sm += expf(row[i] - mx);
    red[tid] = sm; __syncthreads();
    for (int s = 128; s > 0; s >>= 1) { if (tid < s) red[tid] += red[tid + s]; __syncthreads(); }
    const float inv = 1.0f / red[0];

    for (int i = tid; i < V_; i += 256) vocab[(long)b * V_ + i] = expf(row[i] - mx) * inv;
}

// ---------------- Host launcher ----------------
extern "C" void kernel_launch(void* const* d_in, const int* in_sizes, int n_in,
                              void* d_out, int out_size, void* d_ws, size_t ws_size,
                              hipStream_t stream) {
    const float* x      = (const float*)d_in[0];   // [64,512]
    const float* hprev  = (const float*)d_in[1];   // [64,512]
    const float* enc    = (const float*)d_in[2];   // [64,400,1024]
    const unsigned char* mask = (const unsigned char*)d_in[3]; // [64,400] bool
    const float* w_ih   = (const float*)d_in[4];   // [1536,512]
    const float* w_hh   = (const float*)d_in[5];   // [1536,512]
    const float* b_ih   = (const float*)d_in[6];   // [1536]
    const float* b_hh   = (const float*)d_in[7];   // [1536]
    const float* Wh_w   = (const float*)d_in[8];   // [512,1024]
    const float* Wh_b   = (const float*)d_in[9];   // [512]
    const float* Ws_w   = (const float*)d_in[10];  // [512,512]
    const float* Ws_b   = (const float*)d_in[11];  // [512]
    const float* vatt   = (const float*)d_in[12];  // [1,512]
    const float* V_w    = (const float*)d_in[13];  // [512,1536]
    const float* V_b    = (const float*)d_in[14];  // [512]
    const float* Vhat_w = (const float*)d_in[15];  // [32000,512]
    const float* Vhat_b = (const float*)d_in[16];  // [32000]

    // Output regions (tuple order: vocab_dist, attn_dist, context, hidden)
    float* out        = (float*)d_out;
    float* out_vocab  = out;
    float* out_attn   = out_vocab + (long)B_ * V_;
    float* out_ctx    = out_attn + (long)B_ * S_;
    float* out_hidden = out_ctx + (long)B_ * 2 * H_;

    // Workspace layout (floats, then bf16 tail)
    float* ws     = (float*)d_ws;
    float* gi     = ws;                         // 64*1536
    float* gh     = gi + (long)B_ * 3 * H_;     // 64*1536
    float* sproj  = gh + (long)B_ * 3 * H_;     // 64*512
    float* scores = sproj + (long)B_ * H_;      // 64*400
    float* concat = scores + (long)B_ * S_;     // 64*1536
    float* outp   = concat + (long)B_ * 3 * H_; // 64*512
    float* logits = outp + (long)B_ * H_;       // 64*32000
    __bf16* Whbf  = (__bf16*)(logits + (long)B_ * V_); // 512*1024 bf16 (1 MB)

    // 0. One-shot weight conversion for the dominant GEMM's B stream
    f32_to_bf16_kernel<<<(H_ * 2 * H_ + 255) / 256, 256, 0, stream>>>(Wh_w, Whbf, H_ * 2 * H_);
    // 1-2. GRU gate GEMMs: [64,512]@[512,1536] + bias
    gemm_nt_bf16_wmma<<<dim3(12, 4), 256, 0, stream>>>(x,     w_ih, b_ih, gi, B_, 3 * H_, H_);
    gemm_nt_bf16_wmma<<<dim3(12, 4), 256, 0, stream>>>(hprev, w_hh, b_hh, gh, B_, 3 * H_, H_);
    // 3. GRU elementwise -> hidden (also into concat[:, :512])
    gru_cell_kernel<<<(B_ * H_ + 255) / 256, 256, 0, stream>>>(gi, gh, hprev, out_hidden, concat);
    // 4. sproj = hidden @ Ws^T + Ws_b   [64,512]
    gemm_nt_bf16_wmma<<<dim3(4, 4), 256, 0, stream>>>(out_hidden, Ws_w, Ws_b, sproj, B_, H_, H_);
    // 5. Fused big GEMM + tanh + dot(vatt): scores [B,S]   (25600x512x1024, TDM-staged A)
    attn_scores_wmma<<<(B_ * S_) / 16, 256, 0, stream>>>(enc, Whbf, Wh_b, sproj, vatt, scores);
    // 6. Masked softmax over S
    softmax_attn_kernel<<<B_, 256, 0, stream>>>(scores, mask, out_attn);
    // 7. Context vector (also into concat[:, 512:1536])
    context_kernel<<<B_, 256, 0, stream>>>(out_attn, enc, out_ctx, concat);
    // 8. out = concat @ V_w^T + V_b   [64,512], K=1536
    gemm_nt_bf16_wmma<<<dim3(4, 4), 256, 0, stream>>>(concat, V_w, V_b, outp, B_, H_, 3 * H_);
    // 9. logits = out @ Vhat^T + Vhat_b   [64,32000], K=512
    gemm_nt_bf16_wmma<<<dim3(250, 4), 256, 0, stream>>>(outp, Vhat_w, Vhat_b, logits, B_, V_, H_);
    // 10. Vocab softmax
    softmax_vocab_kernel<<<B_, 256, 0, stream>>>(logits, out_vocab);
}
